// BAE_11261404250415
// MI455X (gfx1250) — compile-verified, matrix operand order checked by reading, not run on hardware
//
#include <hip/hip_runtime.h>
#include <hip/hip_bf16.h>

// ---------------------------------------------------------------------------
// Problem constants
// ---------------------------------------------------------------------------
#define BROWS   32768
#define D_IN    1024
#define HID     1024
#define LAT     256
#define OUT_DIM 1050
#define EDGE_DIM 900
#define NODE_DIM 150
#define BN_EPS  1e-5f

typedef __bf16 v16bf __attribute__((ext_vector_type(16)));
typedef float  v8f   __attribute__((ext_vector_type(8)));
typedef int    v4i   __attribute__((ext_vector_type(4)));

union Frag16 { v16bf v; unsigned int u[8]; };

#define AS1 __attribute__((address_space(1)))
#define AS3 __attribute__((address_space(3)))

#if defined(__AMDGCN__) && __has_builtin(__builtin_amdgcn_global_load_async_to_lds_b128)
#define USE_ASYNC_LDS 1
#else
#define USE_ASYNC_LDS 0
#endif

__device__ __forceinline__ AS1 v4i* to_global4(const void* p) {
    return (AS1 v4i*)const_cast<void*>(p);
}
__device__ __forceinline__ AS3 v4i* to_lds4(void* p) {
    return (AS3 v4i*)p;
}

__device__ __forceinline__ void wait_async_all() {
#if __has_builtin(__builtin_amdgcn_s_wait_asynccnt)
    __builtin_amdgcn_s_wait_asynccnt(0);
#else
    asm volatile("s_wait_asynccnt 0x0" ::: "memory");
#endif
}

// f32 -> bf16 bits with round-to-nearest-even
__device__ __forceinline__ unsigned short f2bf(float f) {
    unsigned int u = __float_as_uint(f);
    u += 0x7FFFu + ((u >> 16) & 1u);
    return (unsigned short)(u >> 16);
}

// ---------------------------------------------------------------------------
// Prep kernels: bf16 conversion / weight transpose-conversion
// ---------------------------------------------------------------------------
__global__ __launch_bounds__(256)
void to_bf16_kernel(const float* __restrict__ in, unsigned short* __restrict__ out,
                    long long n4)   // n4 = n/4, n multiple of 4
{
    long long i = (long long)blockIdx.x * 256 + threadIdx.x;
    if (i >= n4) return;
    const float4 f = *(const float4*)(in + i * 4);
    ushort4 o;
    o.x = f2bf(f.x); o.y = f2bf(f.y); o.z = f2bf(f.z); o.w = f2bf(f.w);
    *(ushort4*)(out + i * 4) = o;
}

// W[K][N] f32  ->  Wt[N][K] bf16  (32x32 tiles via LDS, bounds-checked)
__global__ __launch_bounds__(256)
void transpose_to_bf16(const float* __restrict__ W, unsigned short* __restrict__ Wt,
                       int K, int N)
{
    __shared__ float tile[32][33];
    const int kb = blockIdx.y * 32, nb = blockIdx.x * 32;
    const int tx = threadIdx.x & 31, ty = threadIdx.x >> 5;   // ty: 0..7
    #pragma unroll
    for (int i = 0; i < 4; ++i) {
        int kk = ty + i * 8;
        int gk = kb + kk, gn = nb + tx;
        tile[kk][tx] = (gk < K && gn < N) ? W[(size_t)gk * N + gn] : 0.f;
    }
    __syncthreads();
    #pragma unroll
    for (int i = 0; i < 4; ++i) {
        int nn = ty + i * 8;
        int gn = nb + nn, gk = kb + tx;
        if (gn < N && gk < K)
            Wt[(size_t)gn * K + gk] = f2bf(tile[tx][nn]);
    }
}

// ---------------------------------------------------------------------------
// WMMA bf16 GEMM:  C = epilogue(A[M,K](bf16) * Bt[N,K](bf16)^T + bias)
//   RELU_BN : val = g*(relu(val)-mean)*rsqrt(var+eps)+beta
//   WF32    : store f32 to Cf ;  WBF16 : store bf16 to Cb
// Block 256 threads (8 waves), tile 128(M) x 128(N), K-step 32.
// Wave grid 4(M) x 2(N); wave tile 32x64 via 2x4 v_wmma_f32_16x16x32_bf16.
// Staging: GLOBAL_LOAD_ASYNC_TO_LDS_B128 ping-pong (ASYNCcnt), no VGPR staging.
// M, K multiples of 128/32; Bt must have rows padded to a multiple of 128
// (garbage rows only feed discarded ncol >= N output columns).
// ---------------------------------------------------------------------------
#define TM 128
#define TN 128
#define TK 32
#define A_LD 48   // padded LDS row (bf16 units), 96B keeps b128 stores aligned
#define B_LD 48

template <bool RELU_BN, bool WF32, bool WBF16>
__global__ __launch_bounds__(256)
void gemm_bf16_wmma(const unsigned short* __restrict__ A,
                    const unsigned short* __restrict__ Bt,
                    const float* __restrict__ bias,
                    const float* __restrict__ bn_g, const float* __restrict__ bn_b,
                    const float* __restrict__ bn_m, const float* __restrict__ bn_v,
                    float* __restrict__ Cf, unsigned short* __restrict__ Cb,
                    int M, int N, int K, int ldc)
{
    __shared__ unsigned short As[2][TM * A_LD];   // As[.][m][k]
    __shared__ unsigned short Bs[2][TN * B_LD];   // Bs[.][n][k]

    const int tid  = threadIdx.x;
    const int m0   = blockIdx.y * TM;
    const int n0   = blockIdx.x * TN;
    const int w    = tid >> 5;
    const int lane = tid & 31;
    const int wm0  = (w >> 1) * 32;       // wave row offset (4 M-waves)
    const int wn0  = (w & 1) * 64;        // wave col offset (2 N-waves)
    const int ml   = lane & 15;
    const int half = lane >> 4;

    // staging map: tile = 128 rows x 32 bf16 = 512 x b128 chunks, 2/thread
    const int sr = tid >> 2;              // row within tile (+64 on pass 1)
    const int sc = (tid & 3) * 8;         // bf16 col offset

    v8f acc[2][4];
    {
        const v8f vzero = {0.f,0.f,0.f,0.f,0.f,0.f,0.f,0.f};
        #pragma unroll
        for (int mt = 0; mt < 2; ++mt)
            #pragma unroll
            for (int nt = 0; nt < 4; ++nt) acc[mt][nt] = vzero;
    }

#if USE_ASYNC_LDS
    auto stage_tiles = [&](int ks, int buf) {
        #pragma unroll
        for (int p = 0; p < 2; ++p) {
            int row = sr + p * 64;
            __builtin_amdgcn_global_load_async_to_lds_b128(
                to_global4(A + (size_t)(m0 + row) * K + ks + sc),
                to_lds4(&As[buf][row * A_LD + sc]), 0, 0);
            __builtin_amdgcn_global_load_async_to_lds_b128(
                to_global4(Bt + (size_t)(n0 + row) * K + ks + sc),
                to_lds4(&Bs[buf][row * B_LD + sc]), 0, 0);
        }
    };
#else
    uint4 ra[2], rb[2];
    auto load_tiles = [&](int ks) {
        #pragma unroll
        for (int p = 0; p < 2; ++p) {
            ra[p] = *(const uint4*)(A  + (size_t)(m0 + sr + p * 64) * K + ks + sc);
            rb[p] = *(const uint4*)(Bt + (size_t)(n0 + sr + p * 64) * K + ks + sc);
        }
    };
    auto store_tiles = [&](int buf) {
        #pragma unroll
        for (int p = 0; p < 2; ++p) {
            *(uint4*)(&As[buf][(sr + p * 64) * A_LD + sc]) = ra[p];
            *(uint4*)(&Bs[buf][(sr + p * 64) * B_LD + sc]) = rb[p];
        }
    };
#endif

    const int nk = K / TK;
#if USE_ASYNC_LDS
    stage_tiles(0, 0);
    wait_async_all();
#else
    load_tiles(0);
    store_tiles(0);
#endif
    __syncthreads();

    int cur = 0;
    for (int i = 0; i < nk; ++i) {
        const bool has_next = (i + 1) < nk;
#if USE_ASYNC_LDS
        if (has_next) stage_tiles((i + 1) * TK, cur ^ 1);  // overlaps WMMAs
#else
        if (has_next) load_tiles((i + 1) * TK);
#endif

        const unsigned short* Ab = As[cur];
        const unsigned short* Bb = Bs[cur];
        Frag16 fa[2], fb[4];
        #pragma unroll
        for (int mt = 0; mt < 2; ++mt) {
            const unsigned short* p = &Ab[(wm0 + mt * 16 + ml) * A_LD];
            #pragma unroll
            for (int vv = 0; vv < 8; ++vv) {
                int kk = ((vv < 4) ? 0 : 16) + half * 8 + ((vv & 3) << 1);
                fa[mt].u[vv] = *(const unsigned int*)(p + kk);
            }
        }
        #pragma unroll
        for (int nt = 0; nt < 4; ++nt) {
            const unsigned short* p = &Bb[(wn0 + nt * 16 + ml) * B_LD];
            #pragma unroll
            for (int vv = 0; vv < 8; ++vv) {
                int kk = ((vv < 4) ? 0 : 16) + half * 8 + ((vv & 3) << 1);
                fb[nt].u[vv] = *(const unsigned int*)(p + kk);
            }
        }
        #pragma unroll
        for (int mt = 0; mt < 2; ++mt)
            #pragma unroll
            for (int nt = 0; nt < 4; ++nt)
                acc[mt][nt] = __builtin_amdgcn_wmma_f32_16x16x32_bf16(
                    false, fa[mt].v, false, fb[nt].v,
                    (short)0, acc[mt][nt], false, false);

#if USE_ASYNC_LDS
        if (has_next) wait_async_all();
#else
        if (has_next) store_tiles(cur ^ 1);
#endif
        __syncthreads();
        cur ^= 1;
    }

    // epilogue: VGPR r -> M = half*8 + r, N = ml (per 16x16 tile)
    #pragma unroll
    for (int mt = 0; mt < 2; ++mt) {
        #pragma unroll
        for (int nt = 0; nt < 4; ++nt) {
            int ncol = n0 + wn0 + nt * 16 + ml;
            if (ncol >= N) continue;
            float bia = bias[ncol];
            float g = 0.f, bb2 = 0.f, mn = 0.f, inv = 0.f;
            if (RELU_BN) {
                g = bn_g[ncol]; bb2 = bn_b[ncol]; mn = bn_m[ncol];
                inv = rsqrtf(bn_v[ncol] + BN_EPS);
            }
            #pragma unroll
            for (int r = 0; r < 8; ++r) {
                int mrow = m0 + wm0 + mt * 16 + half * 8 + r;
                float val = acc[mt][nt][r] + bia;
                if (RELU_BN) {
                    val = fmaxf(val, 0.f);
                    val = g * (val - mn) * inv + bb2;
                }
                if (WF32)  Cf[(size_t)mrow * ldc + ncol] = val;
                if (WBF16) Cb[(size_t)mrow * ldc + ncol] = f2bf(val);
            }
        }
    }
}

// ---------------------------------------------------------------------------
// Log path: lm = mean_t(log[b,t,:]); a_ln = lm*W[9x12]+b; a_le = lm*W[9x78]+b
// (seq-len-1 attention => softmax weight == 1, q/k are dead)
// ---------------------------------------------------------------------------
__global__ __launch_bounds__(256)
void log_attn_kernel(const float* __restrict__ logx,
                     const float* __restrict__ ln_vw, const float* __restrict__ ln_vb,
                     const float* __restrict__ le_vw, const float* __restrict__ le_vb,
                     float* __restrict__ a_ln, float* __restrict__ a_le)
{
    int b = blockIdx.x * blockDim.x + threadIdx.x;
    if (b >= BROWS) return;
    const float* p = logx + (size_t)b * 288;
    float lm[9];
    #pragma unroll
    for (int d = 0; d < 9; ++d) lm[d] = 0.f;
    for (int t = 0; t < 32; ++t)
        #pragma unroll
        for (int d = 0; d < 9; ++d) lm[d] += p[t * 9 + d];
    #pragma unroll
    for (int d = 0; d < 9; ++d) lm[d] *= (1.f / 32.f);
    for (int j = 0; j < 12; ++j) {
        float s = ln_vb[j];
        #pragma unroll
        for (int d = 0; d < 9; ++d) s += lm[d] * ln_vw[d * 12 + j];
        a_ln[(size_t)b * 12 + j] = s;
    }
    for (int j = 0; j < 78; ++j) {
        float s = le_vb[j];
        #pragma unroll
        for (int d = 0; d < 9; ++d) s += lm[d] * le_vw[d * 78 + j];
        a_le[(size_t)b * 78 + j] = s;
    }
}

// ---------------------------------------------------------------------------
// Final assembly: symmetrize edge, scatter-add, emit edge[B,900], node[B,150]
// ---------------------------------------------------------------------------
__constant__ int c_opt_off[55] = {
    620,621,622,623,624,625,626,627,628,629,
    651,652,653,654,655,656,657,658,659,
    682,683,684,685,686,687,688,689,
    713,714,715,716,717,718,719,
    744,745,746,747,748,749,
    775,776,777,778,779,
    806,807,808,809,
    837,838,839,
    868,869,
    899};
__constant__ int c_log_off[78] = {
    620,621,622,623,624,625,626,627,628,629,
    651,652,653,654,655,656,657,658,659,
    682,683,684,685,686,687,688,689,
    713,714,715,716,717,718,719,
    744,745,746,747,748,749,
    775,776,777,778,779,
    806,807,808,809,
    837,838,839,
    868,869,
    899,
    440,441,442,443,444,445,446,447,448,449,434,435,
    470,471,472,473,474,475,476,477,478,479,465};

__global__ __launch_bounds__(128)
void assemble_kernel(const float* __restrict__ raw,
                     const float* __restrict__ a_on, const float* __restrict__ a_oe,
                     const float* __restrict__ a_ln, const float* __restrict__ a_le,
                     float* __restrict__ out_edge, float* __restrict__ out_node)
{
    const int b = blockIdx.x;
    const int tid = threadIdx.x;
    __shared__ float s_raw[OUT_DIM];
    __shared__ float s_edge[EDGE_DIM];
    __shared__ float s_oe[55], s_le[78], s_on[10], s_ln[12];

    const float* r = raw + (size_t)b * OUT_DIM;
    for (int i = tid; i < OUT_DIM; i += 128) s_raw[i] = r[i];
    for (int i = tid; i < 55; i += 128) s_oe[i] = a_oe[(size_t)b * 55 + i];
    for (int i = tid; i < 78; i += 128) s_le[i] = a_le[(size_t)b * 78 + i];
    if (tid < 10) s_on[tid] = a_on[(size_t)b * 10 + tid];
    if (tid < 12) s_ln[tid] = a_ln[(size_t)b * 12 + tid];
    __syncthreads();

    for (int e = tid; e < EDGE_DIM; e += 128) {
        int i = e / 30, j = e % 30;
        s_edge[e] = 0.5f * (s_raw[i * 30 + j] + s_raw[j * 30 + i]);
    }
    __syncthreads();
    if (tid < 55) s_edge[c_opt_off[tid]] += s_oe[tid];
    __syncthreads();
    if (tid < 78) s_edge[c_log_off[tid]] += s_le[tid];
    __syncthreads();

    float* oe = out_edge + (size_t)b * EDGE_DIM;
    for (int e = tid; e < EDGE_DIM; e += 128) oe[e] = s_edge[e];

    float* on = out_node + (size_t)b * NODE_DIM;
    for (int idx = tid; idx < NODE_DIM; idx += 128) {
        int i = idx / 5;
        float v = s_raw[EDGE_DIM + idx];
        if (i >= 20)       v += s_on[i - 20] + s_ln[i - 20];
        else if (i == 14)  v += s_ln[10];
        else if (i == 15)  v += s_ln[11];
        on[idx] = v;
    }
}

// ---------------------------------------------------------------------------
// Launch
// ---------------------------------------------------------------------------
extern "C" void kernel_launch(void* const* d_in, const int* in_sizes, int n_in,
                              void* d_out, int out_size, void* d_ws, size_t ws_size,
                              hipStream_t stream)
{
    (void)in_sizes; (void)n_in; (void)out_size; (void)ws_size;

    const float* x       = (const float*)d_in[0];
    const float* optical = (const float*)d_in[1];
    const float* logx    = (const float*)d_in[2];
    const float* enc_w   = (const float*)d_in[3];
    const float* enc_b   = (const float*)d_in[4];
    const float* enc_g   = (const float*)d_in[5];
    const float* enc_bb  = (const float*)d_in[6];
    const float* enc_m   = (const float*)d_in[7];
    const float* enc_v   = (const float*)d_in[8];
    const float* mu_w    = (const float*)d_in[9];
    const float* mu_b    = (const float*)d_in[10];
    const float* lv_w    = (const float*)d_in[11];
    const float* lv_b    = (const float*)d_in[12];
    const float* dec1_w  = (const float*)d_in[13];
    const float* dec1_b  = (const float*)d_in[14];
    const float* dec_g   = (const float*)d_in[15];
    const float* dec_bb  = (const float*)d_in[16];
    const float* dec_m   = (const float*)d_in[17];
    const float* dec_v   = (const float*)d_in[18];
    const float* dec2_w  = (const float*)d_in[19];
    const float* dec2_b  = (const float*)d_in[20];
    // q/k weights (21-24, 27-30, 33-36, 39-42) are dead: softmax over a
    // singleton axis == 1 exactly, so attention output == v-projection.
    const float* on_vw   = (const float*)d_in[25];
    const float* on_vb   = (const float*)d_in[26];
    const float* oe_vw   = (const float*)d_in[31];
    const float* oe_vb   = (const float*)d_in[32];
    const float* ln_vw   = (const float*)d_in[37];
    const float* ln_vb   = (const float*)d_in[38];
    const float* le_vw   = (const float*)d_in[43];
    const float* le_vb   = (const float*)d_in[44];

    // output layout: edge[B*900] | node[B*150] | mu[B*256] | logvar[B*256]
    float* out_edge = (float*)d_out;
    float* out_node = out_edge + (size_t)BROWS * EDGE_DIM;
    float* out_mu   = out_node + (size_t)BROWS * NODE_DIM;
    float* out_lv   = out_mu   + (size_t)BROWS * LAT;

    // workspace layout; weight row-counts padded to multiples of TN=128 so
    // async B-tile loads are always in-bounds.
    const int NP_MU  = 256, NP_HID = 1024, NP_DEC2 = 1152, NP_SMALL = 128;
    char* c = (char*)d_ws;
    unsigned short* xb     = (unsigned short*)c; c += (size_t)BROWS * D_IN * 2;
    unsigned short* ob     = (unsigned short*)c; c += (size_t)BROWS * D_IN * 2;
    unsigned short* h_bf   = (unsigned short*)c; c += (size_t)BROWS * HID * 2;
    unsigned short* mu_bf  = (unsigned short*)c; c += (size_t)BROWS * LAT * 2;
    unsigned short* h2_bf  = (unsigned short*)c; c += (size_t)BROWS * HID * 2;
    float* raw  = (float*)c; c += (size_t)BROWS * OUT_DIM * 4;
    float* a_on = (float*)c; c += (size_t)BROWS * 10 * 4;
    float* a_oe = (float*)c; c += (size_t)BROWS * 55 * 4;
    float* a_ln = (float*)c; c += (size_t)BROWS * 12 * 4;
    float* a_le = (float*)c; c += (size_t)BROWS * 78 * 4;
    unsigned short* wt_enc  = (unsigned short*)c; c += (size_t)NP_HID  * D_IN * 2;
    unsigned short* wt_mu   = (unsigned short*)c; c += (size_t)NP_MU   * HID  * 2;
    unsigned short* wt_lv   = (unsigned short*)c; c += (size_t)NP_MU   * HID  * 2;
    unsigned short* wt_dec1 = (unsigned short*)c; c += (size_t)NP_HID  * LAT  * 2;
    unsigned short* wt_dec2 = (unsigned short*)c; c += (size_t)NP_DEC2 * HID  * 2;
    unsigned short* wt_on   = (unsigned short*)c; c += (size_t)NP_SMALL * D_IN * 2;
    unsigned short* wt_oe   = (unsigned short*)c; c += (size_t)NP_SMALL * D_IN * 2;

    const dim3 blk(256);
    const int MB = BROWS / TM;  // 256

    // ---- prep: bf16 conversions ----
    {
        long long n4 = (long long)BROWS * D_IN / 4;
        int nb = (int)((n4 + 255) / 256);
        to_bf16_kernel<<<dim3(nb), blk, 0, stream>>>(x, xb, n4);
        to_bf16_kernel<<<dim3(nb), blk, 0, stream>>>(optical, ob, n4);
    }
    transpose_to_bf16<<<dim3(HID/32,   D_IN/32), blk, 0, stream>>>(enc_w,  wt_enc,  D_IN, HID);
    transpose_to_bf16<<<dim3(LAT/32,   HID/32),  blk, 0, stream>>>(mu_w,   wt_mu,   HID,  LAT);
    transpose_to_bf16<<<dim3(LAT/32,   HID/32),  blk, 0, stream>>>(lv_w,   wt_lv,   HID,  LAT);
    transpose_to_bf16<<<dim3(HID/32,   LAT/32),  blk, 0, stream>>>(dec1_w, wt_dec1, LAT,  HID);
    transpose_to_bf16<<<dim3((OUT_DIM+31)/32, HID/32), blk, 0, stream>>>(dec2_w, wt_dec2, HID, OUT_DIM);
    transpose_to_bf16<<<dim3(1,        D_IN/32), blk, 0, stream>>>(on_vw,  wt_on,   D_IN, 10);
    transpose_to_bf16<<<dim3(2,        D_IN/32), blk, 0, stream>>>(oe_vw,  wt_oe,   D_IN, 55);

    // ---- GEMM chain (v_wmma_f32_16x16x32_bf16 + async-LDS staging) ----
    // 1) h = BN(relu(x @ enc_w + enc_b))        -> bf16
    gemm_bf16_wmma<true,  false, true><<<dim3(HID / TN, MB), blk, 0, stream>>>(
        xb, wt_enc, enc_b, enc_g, enc_bb, enc_m, enc_v,
        nullptr, h_bf, BROWS, HID, D_IN, HID);
    // 2) mu = h @ mu_w + mu_b                   -> f32 output + bf16 for dec1
    gemm_bf16_wmma<false, true,  true><<<dim3(LAT / TN, MB), blk, 0, stream>>>(
        h_bf, wt_mu, mu_b, nullptr, nullptr, nullptr, nullptr,
        out_mu, mu_bf, BROWS, LAT, HID, LAT);
    // 3) logvar = h @ lv_w + lv_b               -> f32 output
    gemm_bf16_wmma<false, true,  false><<<dim3(LAT / TN, MB), blk, 0, stream>>>(
        h_bf, wt_lv, lv_b, nullptr, nullptr, nullptr, nullptr,
        out_lv, nullptr, BROWS, LAT, HID, LAT);
    // 4) h2 = BN(relu(mu @ dec1_w + dec1_b))    -> bf16 (z == mu in eval mode)
    gemm_bf16_wmma<true,  false, true><<<dim3(HID / TN, MB), blk, 0, stream>>>(
        mu_bf, wt_dec1, dec1_b, dec_g, dec_bb, dec_m, dec_v,
        nullptr, h2_bf, BROWS, HID, LAT, HID);
    // 5) raw = h2 @ dec2_w + dec2_b             -> f32
    gemm_bf16_wmma<false, true,  false><<<dim3((OUT_DIM + TN - 1) / TN, MB), blk, 0, stream>>>(
        h2_bf, wt_dec2, dec2_b, nullptr, nullptr, nullptr, nullptr,
        raw, nullptr, BROWS, OUT_DIM, HID, OUT_DIM);
    // 6) a_on = optical @ on_vw + on_vb
    gemm_bf16_wmma<false, true,  false><<<dim3(1, MB), blk, 0, stream>>>(
        ob, wt_on, on_vb, nullptr, nullptr, nullptr, nullptr,
        a_on, nullptr, BROWS, 10, D_IN, 10);
    // 7) a_oe = optical @ oe_vw + oe_vb
    gemm_bf16_wmma<false, true,  false><<<dim3(1, MB), blk, 0, stream>>>(
        ob, wt_oe, oe_vb, nullptr, nullptr, nullptr, nullptr,
        a_oe, nullptr, BROWS, 55, D_IN, 55);

    // 8) log-mean + tiny K=9 matmuls
    log_attn_kernel<<<dim3((BROWS + 255) / 256), blk, 0, stream>>>(
        logx, ln_vw, ln_vb, le_vw, le_vb, a_ln, a_le);
    // 9) symmetrize + scatter-add + emit edge/node
    assemble_kernel<<<dim3(BROWS), dim3(128), 0, stream>>>(
        raw, a_on, a_oe, a_ln, a_le, out_edge, out_node);
}
